// EuclideanCodebook_49271864820005
// MI455X (gfx1250) — compile-verified, matrix operand order checked by reading, not run on hardware
//
#include <hip/hip_runtime.h>
#include <hip/hip_bf16.h>

// Problem constants (fixed by the reference).
#define BT   65536      // 8 * 8192 query rows
#define CCH  128        // channels
#define KCB  1024       // codebook entries
#define NTILE (KCB / 16)        // 64 code tiles of 16
#define FRAG_ELEMS (KCB * CCH)  // 131072 bf16 elements in fragment layout
#define LDS_BYTES (FRAG_ELEMS * 2 + KCB * 4)  // 256KB codebook + 4KB e_sq = 260KB

typedef __attribute__((ext_vector_type(16))) __bf16    v16bf;
typedef __attribute__((ext_vector_type(8)))  float     v8f;
typedef __attribute__((ext_vector_type(8)))  unsigned  v8u;

static __device__ __forceinline__ __bf16 f2bf(float f) {
    union { float f; unsigned u; } v; v.f = f;
    unsigned r = v.u + 0x7FFFu + ((v.u >> 16) & 1u);   // round-to-nearest-even
    unsigned short h = (unsigned short)(r >> 16);
    union { unsigned short s; __bf16 b; } o; o.s = h;
    return o.b;
}

// Prep: scatter f32 codebook into bf16 WMMA A-fragment layout + compute ||e||^2.
// A (16-bit, 16x32) per-lane layout: lane group g (lanes 16..31 => g=1):
//   pos p<8  -> kk = p + 8g ; pos p>=8 -> kk = p + 8 + 8g
// LDS layout is half-split for bank-conflict-free ds_load_b128:
//   element (tile,chunk,lane,p) -> ((((tile*4+c)*2 + (p>>3))*32 + lane) << 3) + (p&7)
__global__ __launch_bounds__(256)
void vq_prep_kernel(const float* __restrict__ embed,
                    unsigned short* __restrict__ frag,  // bf16 bits
                    float* __restrict__ esq) {
    int i = blockIdx.x * 256 + threadIdx.x;
    if (i < FRAG_ELEMS) {
        int k  = i >> 7;          // code index
        int ch = i & 127;         // channel
        float f = embed[i];
        int tile = k >> 4, m = k & 15;
        int c  = ch >> 5, kk = ch & 31;
        int g  = (kk >> 3) & 1;
        int p  = (kk & 7) + ((kk & 16) ? 8 : 0);
        int lane = m + (g << 4);
        union { __bf16 b; unsigned short s; } o; o.b = f2bf(f);
        frag[(((((tile * 4 + c) * 2) + (p >> 3)) * 32 + lane) << 3) + (p & 7)] = o.s;
    }
    if (i < KCB) {
        const float* e = embed + (size_t)i * CCH;
        float s = 0.f;
        #pragma unroll 8
        for (int ch = 0; ch < CCH; ++ch) s += e[ch] * e[ch];
        esq[i] = s;
    }
}

// Main: 256 blocks x 512 threads (16 waves -> 4 waves/SIMD under 260KB LDS).
// Block stages the full bf16 codebook + e_sq into LDS; each wave owns 16
// x-rows (B fragments in registers) and scans all 1024 codes with
// V_WMMA_F32_16X16X32_BF16. argmax(2*dot - e_sq) per row, then gathers.
__global__ __launch_bounds__(512)
void vq_main_kernel(const float* __restrict__ x,
                    const float* __restrict__ embed,
                    const unsigned short* __restrict__ frag,
                    const float* __restrict__ esq,
                    float* __restrict__ quant,
                    int* __restrict__ ind) {
    extern __shared__ char smem[];
    const int tid = threadIdx.x;

    // ---- stage codebook fragments + e_sq into LDS (coalesced 16B copies) ----
    {
        const uint4* src = (const uint4*)frag;            // 16384 x 16B
        uint4* dst = (uint4*)smem;
        for (int i = tid; i < FRAG_ELEMS / 8; i += 512) dst[i] = src[i];
        float* sq = (float*)(smem + FRAG_ELEMS * 2);
        for (int i = tid; i < KCB; i += 512) sq[i] = esq[i];
    }
    __syncthreads();

    const uint4*  sE4 = (const uint4*)smem;               // fragment halves
    const float*  sQ  = (const float*)(smem + FRAG_ELEMS * 2);

    const int wave = tid >> 5;
    const int lane = tid & 31;
    const int g    = lane >> 4;       // half-wave group
    const int n    = lane & 15;       // column (x-row) within tile
    const int rowBase = blockIdx.x * 256 + wave * 16;
    const int row     = rowBase + n;

    // ---- load this lane's x-row channels as 4 bf16 B-fragments ----
    const float* xr = x + (size_t)row * CCH;
    v16bf bfrag[4];
    #pragma unroll
    for (int c = 0; c < 4; ++c) {
        const int base = 32 * c + 8 * g;
        float4 q0 = *(const float4*)(xr + base);
        float4 q1 = *(const float4*)(xr + base + 4);
        float4 q2 = *(const float4*)(xr + base + 16);
        float4 q3 = *(const float4*)(xr + base + 20);
        v16bf b;
        b[0]=f2bf(q0.x);  b[1]=f2bf(q0.y);  b[2]=f2bf(q0.z);  b[3]=f2bf(q0.w);
        b[4]=f2bf(q1.x);  b[5]=f2bf(q1.y);  b[6]=f2bf(q1.z);  b[7]=f2bf(q1.w);
        b[8]=f2bf(q2.x);  b[9]=f2bf(q2.y);  b[10]=f2bf(q2.z); b[11]=f2bf(q2.w);
        b[12]=f2bf(q3.x); b[13]=f2bf(q3.y); b[14]=f2bf(q3.z); b[15]=f2bf(q3.w);
        bfrag[c] = b;
    }

    // ---- scan all code tiles ----
    float best = -3.402823466e38f;
    int bestIdx = 0;
    for (int t = 0; t < NTILE; ++t) {
        v8f acc = {};
        #pragma unroll
        for (int c = 0; c < 4; ++c) {
            // two bank-conflict-free 16B half loads, assembled into one fragment
            const int hbase = (t * 4 + c) * 64 + lane;
            uint4 h0 = sE4[hbase];
            uint4 h1 = sE4[hbase + 32];
            v8u u; u[0]=h0.x; u[1]=h0.y; u[2]=h0.z; u[3]=h0.w;
                   u[4]=h1.x; u[5]=h1.y; u[6]=h1.z; u[7]=h1.w;
            v16bf afrag = __builtin_bit_cast(v16bf, u);
            acc = __builtin_amdgcn_wmma_f32_16x16x32_bf16(
                      false, afrag, false, bfrag[c], (short)0, acc, false, false);
        }
        const int cb = t * 16 + 8 * g;   // this lane's first code in tile
        v8f eq = *(const v8f*)(sQ + cb); // 32B broadcast within half-wave
        float s0 = __builtin_fmaf(acc[0], 2.0f, -eq[0]);
        float s1 = __builtin_fmaf(acc[1], 2.0f, -eq[1]);
        float s2 = __builtin_fmaf(acc[2], 2.0f, -eq[2]);
        float s3 = __builtin_fmaf(acc[3], 2.0f, -eq[3]);
        float s4 = __builtin_fmaf(acc[4], 2.0f, -eq[4]);
        float s5 = __builtin_fmaf(acc[5], 2.0f, -eq[5]);
        float s6 = __builtin_fmaf(acc[6], 2.0f, -eq[6]);
        float s7 = __builtin_fmaf(acc[7], 2.0f, -eq[7]);
        // balanced argmax tree; strict '>' keeps the lower index on ties
        float a01 = s0; int i01 = 0; if (s1 > a01) { a01 = s1; i01 = 1; }
        float a23 = s2; int i23 = 2; if (s3 > a23) { a23 = s3; i23 = 3; }
        float a45 = s4; int i45 = 4; if (s5 > a45) { a45 = s5; i45 = 5; }
        float a67 = s6; int i67 = 6; if (s7 > a67) { a67 = s7; i67 = 7; }
        float a03 = a01; int i03 = i01; if (a23 > a03) { a03 = a23; i03 = i23; }
        float a47 = a45; int i47 = i45; if (a67 > a47) { a47 = a67; i47 = i67; }
        float at  = a03; int it  = i03; if (a47 > at)  { at  = a47; it  = i47; }
        if (at > best) { best = at; bestIdx = cb + it; }
    }

    // ---- combine the two half-wave code ranges (prefer lower index on tie) ----
    float obest = __shfl_xor(best, 16, 32);
    int   oidx  = __shfl_xor(bestIdx, 16, 32);
    if (obest > best || (obest == best && oidx < bestIdx)) { best = obest; bestIdx = oidx; }

    if (lane < 16) ind[row] = bestIdx;

    // ---- gather winning code rows: 32 lanes x float4 = 128 floats per row ----
    #pragma unroll 1
    for (int r = 0; r < 16; ++r) {
        int idx = __shfl(bestIdx, r, 32);
        const float4* src = (const float4*)(embed + (size_t)idx * CCH);
        float4* dst = (float4*)(quant + (size_t)(rowBase + r) * CCH);
        dst[lane] = src[lane];
    }
}

extern "C" void kernel_launch(void* const* d_in, const int* in_sizes, int n_in,
                              void* d_out, int out_size, void* d_ws, size_t ws_size,
                              hipStream_t stream) {
    const float* x     = (const float*)d_in[0];   // [8,8192,128] f32
    const float* embed = (const float*)d_in[1];   // [1024,128] f32

    float* quant = (float*)d_out;                            // [BT,128] f32
    int*   ind   = (int*)((float*)d_out + (size_t)BT * CCH); // [BT] int32

    unsigned short* frag = (unsigned short*)d_ws;                 // 256KB bf16 fragments
    float* esq = (float*)((char*)d_ws + (size_t)FRAG_ELEMS * 2);  // 4KB ||e||^2

    vq_prep_kernel<<<(FRAG_ELEMS + 255) / 256, 256, 0, stream>>>(embed, frag, esq);
    vq_main_kernel<<<BT / 256, 512, LDS_BYTES, stream>>>(x, embed, frag, esq, quant, ind);
}
